// SwinTransformerBlock_35966056137225
// MI455X (gfx1250) — matrix operand order repaired and checked by reading, not compiled, not run on hardware
//
#include <hip/hip_runtime.h>
#include <hip/hip_bf16.h>
#include <math.h>

typedef __attribute__((ext_vector_type(16))) _Float16 v16h;
typedef __attribute__((ext_vector_type(8)))  float    v8f;

union FragU { uint4 u[2]; v16h h; };

__device__ __forceinline__ v8f wmma_f16(v16h a, v16h b, v8f c) {
  // D = A(16x32 f16) x B(32x16 f16) + C(16x16 f32)
  return __builtin_amdgcn_wmma_f32_16x16x32_f16(false, a, false, b, (short)0, c,
                                                false, false);
}

__device__ __forceinline__ v8f zero8() {
  v8f z = {0.f, 0.f, 0.f, 0.f, 0.f, 0.f, 0.f, 0.f};
  return z;
}

// Load a v16h fragment as two 16-byte LDS reads at word offsets wlo / whi
// from a row base pointer.
__device__ __forceinline__ v16h ldsFrag(const _Float16* p, int wlo, int whi) {
  const unsigned* rw = (const unsigned*)p;
  FragU f;
  f.u[0] = *(const uint4*)(rw + wlo);
  f.u[1] = *(const uint4*)(rw + whi);
  return f.h;
}

// ---------------- weight fp32 [K,N] -> f16 [N,K] ----------------
__global__ __launch_bounds__(256) void wt_f16_kernel(
    const float* __restrict__ w, _Float16* __restrict__ wt, int K, int N) {
  int idx = blockIdx.x * 256 + threadIdx.x;
  if (idx >= K * N) return;
  int n = idx / K, k = idx - n * K;
  wt[idx] = (_Float16)w[(size_t)k * N + n];
}

// ---------------- LayerNorm fp32 -> f16, C=256, one wave per token ----------
__global__ __launch_bounds__(256) void ln_f16_kernel(
    const float* __restrict__ x, const float* __restrict__ g,
    const float* __restrict__ b, _Float16* __restrict__ out) {
  int wave = threadIdx.x >> 5, lane = threadIdx.x & 31;
  size_t tok = (size_t)blockIdx.x * 8 + wave;
  const float* row = x + tok * 256 + lane * 8;
  float4 a0 = *(const float4*)row;
  float4 a1 = *(const float4*)(row + 4);
  float v[8] = {a0.x, a0.y, a0.z, a0.w, a1.x, a1.y, a1.z, a1.w};
  float s = 0.f;
#pragma unroll
  for (int i = 0; i < 8; i++) s += v[i];
#pragma unroll
  for (int m = 1; m < 32; m <<= 1) s += __shfl_xor(s, m, 32);
  float mean = s * (1.0f / 256.0f);
  float vs = 0.f;
#pragma unroll
  for (int i = 0; i < 8; i++) { float d = v[i] - mean; vs += d * d; }
#pragma unroll
  for (int m = 1; m < 32; m <<= 1) vs += __shfl_xor(vs, m, 32);
  float inv = rsqrtf(vs * (1.0f / 256.0f) + 1e-5f);
  union { _Float16 hh[8]; uint4 u; } pk;
#pragma unroll
  for (int i = 0; i < 8; i++) {
    int c = lane * 8 + i;
    pk.hh[i] = (_Float16)((v[i] - mean) * inv * g[c] + b[c]);
  }
  *(uint4*)(out + tok * 256 + lane * 8) = pk.u;
}

// ---------------- Generic WMMA GEMM: out = A[M,K](f16) * BT[N,K](f16)^T -----
// mode 0: outH = f16(acc + bias)
// mode 1: outF = acc + bias + res            (fp32)
// mode 2: outH = f16(gelu_erf(acc + bias))
__global__ __launch_bounds__(256) void gemm_wmma_kernel(
    const _Float16* __restrict__ A, const _Float16* __restrict__ BT,
    const float* __restrict__ bias, const float* __restrict__ res,
    _Float16* __restrict__ outH, float* __restrict__ outF,
    int M, int N, int K, int mode) {
  __shared__ __align__(16) _Float16 As[128][32];
  __shared__ __align__(16) _Float16 Bs[128][32];
  int tid = threadIdx.x;
  int lane = tid & 31, wave = tid >> 5;
  int m0 = blockIdx.y * 128, n0 = blockIdx.x * 128;
  int wm = (wave >> 2) * 64;  // wave M base in block tile
  int wn = (wave & 3) * 32;   // wave N base in block tile
  v8f acc[4][2];
#pragma unroll
  for (int i = 0; i < 4; i++)
#pragma unroll
    for (int j = 0; j < 2; j++) acc[i][j] = zero8();

  int rowL = tid >> 1;          // 0..127
  int colL = (tid & 1) * 16;    // halves
  int aw0 = (lane < 16) ? 0 : 4;   // A-fragment word base
  int bh0 = (lane < 16) ? 0 : 16;  // B-fragment half base

  for (int kc = 0; kc < K; kc += 32) {
    const uint4* gA = (const uint4*)(A + (size_t)(m0 + rowL) * K + kc + colL);
    const uint4* gB = (const uint4*)(BT + (size_t)(n0 + rowL) * K + kc + colL);
    uint4 a0 = gA[0], a1 = gA[1];
    uint4 b0 = gB[0], b1 = gB[1];
    __syncthreads();
    *(uint4*)&As[rowL][colL] = a0;
    *(uint4*)&As[rowL][colL + 8] = a1;
    *(uint4*)&Bs[rowL][colL] = b0;
    *(uint4*)&Bs[rowL][colL + 8] = b1;
    __syncthreads();
    v16h bf[2];
#pragma unroll
    for (int j = 0; j < 2; j++)
      bf[j] = ldsFrag(&Bs[wn + j * 16 + (lane & 15)][0], aw0, aw0 + 8);
#pragma unroll
    for (int i = 0; i < 4; i++) {
      v16h af = ldsFrag(&As[wm + i * 16 + (lane & 15)][0], aw0, aw0 + 8);
#pragma unroll
      for (int j = 0; j < 2; j++) acc[i][j] = wmma_f16(af, bf[j], acc[i][j]);
    }
  }

#pragma unroll
  for (int i = 0; i < 4; i++) {
    int mBase = m0 + wm + i * 16 + ((lane >= 16) ? 8 : 0);
#pragma unroll
    for (int j = 0; j < 2; j++) {
      int n = n0 + wn + j * 16 + (lane & 15);
      float bv = bias[n];
#pragma unroll
      for (int r = 0; r < 8; r++) {
        size_t idx = (size_t)(mBase + r) * N + n;
        float vv = acc[i][j][r] + bv;
        if (mode == 0) {
          outH[idx] = (_Float16)vv;
        } else if (mode == 1) {
          outF[idx] = vv + res[idx];
        } else {
          float ge = 0.5f * vv * (1.0f + erff(vv * 0.70710678118654752f));
          outH[idx] = (_Float16)ge;
        }
      }
    }
  }
}

// ---------------- Windowed attention: 1 block per window, 1 wave per head ---
__global__ __launch_bounds__(256) void attn_wmma_kernel(
    const _Float16* __restrict__ qkv, const float* __restrict__ btab,
    _Float16* __restrict__ o16) {
  __shared__ __align__(16) _Float16 poolQKP[8][4096];  // per head: Q|K, later P
  __shared__ __align__(16) _Float16 poolVT[8][2048];   // per head: V^T [32][64]
  int h = threadIdx.x >> 5, lane = threadIdx.x & 31;
  int bw = blockIdx.x;
  int b = bw / 576, rem = bw % 576, wy = rem / 24, wx = rem % 24;
  _Float16* Q = &poolQKP[h][0];
  _Float16* Km = &poolQKP[h][2048];
  _Float16* P = &poolQKP[h][0];  // aliases Q|K after they are consumed
  _Float16* VT = &poolVT[h][0];

  // gather this head's Q,K,V rows (shifted-window token mapping, shift=4)
  for (int t = 0; t < 64; t++) {
    int i = t >> 3, j = t & 7;
    int hh = wy * 8 + i + 4; if (hh >= 192) hh -= 192;
    int ww = wx * 8 + j + 4; if (ww >= 192) ww -= 192;
    size_t tok = (size_t)b * 36864 + (size_t)hh * 192 + ww;
    const _Float16* base = qkv + tok * 768 + h * 32;
    Q[t * 32 + lane] = base[lane];
    Km[t * 32 + lane] = base[256 + lane];
    VT[lane * 64 + t] = base[512 + lane];
  }
  // LDS here is wave-private (indexed by h): in-wave DS ordering suffices.

  int aw0 = (lane < 16) ? 0 : 4;
  int bh0 = (lane < 16) ? 0 : 16;

  // S = Q @ K^T (hd=32 => single K-step per tile), 4x4 tiles of 16x16
  v8f S[4][4];
#pragma unroll
  for (int mi = 0; mi < 4; mi++)
#pragma unroll
    for (int nj = 0; nj < 4; nj++) S[mi][nj] = zero8();
  v16h kf[4];
#pragma unroll
  for (int nj = 0; nj < 4; nj++)
    kf[nj] = ldsFrag(&Km[(nj * 16 + (lane & 15)) * 32], aw0, aw0 + 8);
#pragma unroll
  for (int mi = 0; mi < 4; mi++) {
    v16h qf = ldsFrag(&Q[(mi * 16 + (lane & 15)) * 32], aw0, aw0 + 8);
#pragma unroll
    for (int nj = 0; nj < 4; nj++) S[mi][nj] = wmma_f16(qf, kf[nj], S[mi][nj]);
  }

  // softmax over rows: scale + rel-pos bias, reduce across 16-lane row groups
  const float scale = 0.17677669529663687f;  // 1/sqrt(32)
#pragma unroll
  for (int mi = 0; mi < 4; mi++) {
#pragma unroll
    for (int r = 0; r < 8; r++) {
      int qrow = mi * 16 + r + ((lane >= 16) ? 8 : 0);
      int qy = qrow >> 3, qx = qrow & 7;
      float vals[4];
      float mx = -3.0e38f;
#pragma unroll
      for (int nj = 0; nj < 4; nj++) {
        int krow = nj * 16 + (lane & 15);
        int ky = krow >> 3, kx = krow & 7;
        int rel = (qy - ky + 7) * 15 + (qx - kx + 7);
        float vv = S[mi][nj][r] * scale + btab[rel * 8 + h];
        vals[nj] = vv;
        mx = fmaxf(mx, vv);
      }
#pragma unroll
      for (int m = 1; m < 16; m <<= 1) mx = fmaxf(mx, __shfl_xor(mx, m, 32));
      float sum = 0.f;
#pragma unroll
      for (int nj = 0; nj < 4; nj++) {
        vals[nj] = __expf(vals[nj] - mx);
        sum += vals[nj];
      }
#pragma unroll
      for (int m = 1; m < 16; m <<= 1) sum += __shfl_xor(sum, m, 32);
      float inv = 1.0f / sum;
#pragma unroll
      for (int nj = 0; nj < 4; nj++) S[mi][nj][r] = vals[nj] * inv;
    }
  }

  // write P (f16) into LDS, overwriting the now-dead Q|K region
#pragma unroll
  for (int mi = 0; mi < 4; mi++)
#pragma unroll
    for (int nj = 0; nj < 4; nj++)
#pragma unroll
      for (int r = 0; r < 8; r++) {
        int m = mi * 16 + r + ((lane >= 16) ? 8 : 0);
        int n = nj * 16 + (lane & 15);
        P[m * 64 + n] = (_Float16)S[mi][nj][r];
      }

  // O = P(64x64) @ V(64x32): 4x2 tiles, K=64 in two 32-chunks
  v8f O[4][2];
#pragma unroll
  for (int mi = 0; mi < 4; mi++)
#pragma unroll
    for (int dj = 0; dj < 2; dj++) O[mi][dj] = zero8();
  v16h vf[2][2];  // [kchunk][dj]
#pragma unroll
  for (int kk = 0; kk < 2; kk++)
#pragma unroll
    for (int dj = 0; dj < 2; dj++)
      vf[kk][dj] =
          ldsFrag(&VT[(dj * 16 + (lane & 15)) * 64] + kk * 32, aw0, aw0 + 8);
#pragma unroll
  for (int mi = 0; mi < 4; mi++) {
#pragma unroll
    for (int kk = 0; kk < 2; kk++) {
      v16h pf = ldsFrag(&P[(mi * 16 + (lane & 15)) * 64] + kk * 32, aw0, aw0 + 8);
#pragma unroll
      for (int dj = 0; dj < 2; dj++)
        O[mi][dj] = wmma_f16(pf, vf[kk][dj], O[mi][dj]);
    }
  }

  // scatter O back to token order (inverse roll folded into mapping)
#pragma unroll
  for (int mi = 0; mi < 4; mi++) {
#pragma unroll
    for (int r = 0; r < 8; r++) {
      int t = mi * 16 + r + ((lane >= 16) ? 8 : 0);
      int i = t >> 3, j = t & 7;
      int hh = wy * 8 + i + 4; if (hh >= 192) hh -= 192;
      int ww = wx * 8 + j + 4; if (ww >= 192) ww -= 192;
      size_t tok = (size_t)b * 36864 + (size_t)hh * 192 + ww;
#pragma unroll
      for (int dj = 0; dj < 2; dj++) {
        int d = dj * 16 + (lane & 15);
        o16[tok * 256 + h * 32 + d] = (_Float16)O[mi][dj][r];
      }
    }
  }
}

extern "C" void kernel_launch(void* const* d_in, const int* in_sizes, int n_in,
                              void* d_out, int out_size, void* d_ws,
                              size_t ws_size, hipStream_t stream) {
  const float* x      = (const float*)d_in[0];
  const float* n1g    = (const float*)d_in[1];
  const float* n1b    = (const float*)d_in[2];
  const float* qkv_w  = (const float*)d_in[3];
  const float* qkv_b  = (const float*)d_in[4];
  const float* btab   = (const float*)d_in[5];
  const float* proj_w = (const float*)d_in[6];
  const float* proj_b = (const float*)d_in[7];
  const float* n2g    = (const float*)d_in[8];
  const float* n2b    = (const float*)d_in[9];
  const float* fc1_w  = (const float*)d_in[10];
  const float* fc1_b  = (const float*)d_in[11];
  const float* fc2_w  = (const float*)d_in[12];
  const float* fc2_b  = (const float*)d_in[13];
  float* out = (float*)d_out;
  (void)in_sizes; (void)n_in; (void)out_size; (void)ws_size;

  const size_t Ntok = 147456;  // 4 * 192 * 192
  char* ws = (char*)d_ws;
  size_t off = 0;
  auto take = [&](size_t bytes) {
    char* p = ws + off;
    off = (off + bytes + 255) & ~(size_t)255;
    return p;
  };
  _Float16* qkvT = (_Float16*)take((size_t)768 * 256 * 2);
  _Float16* projT = (_Float16*)take((size_t)256 * 256 * 2);
  _Float16* fc1T = (_Float16*)take((size_t)1024 * 256 * 2);
  _Float16* fc2T = (_Float16*)take((size_t)256 * 1024 * 2);
  _Float16* h16 = (_Float16*)take(Ntok * 256 * 2);
  _Float16* big = (_Float16*)take(Ntok * 1024 * 2);  // qkv(768)+o(256) = fc1out(1024)
  _Float16* qkvbuf = big;
  _Float16* o16 = big + Ntok * 768;
  _Float16* fc1out = big;

  // one-time weight transposes to f16 [N][K]
  wt_f16_kernel<<<(768 * 256 + 255) / 256, 256, 0, stream>>>(qkv_w, qkvT, 256, 768);
  wt_f16_kernel<<<(256 * 256 + 255) / 256, 256, 0, stream>>>(proj_w, projT, 256, 256);
  wt_f16_kernel<<<(256 * 1024 + 255) / 256, 256, 0, stream>>>(fc1_w, fc1T, 256, 1024);
  wt_f16_kernel<<<(1024 * 256 + 255) / 256, 256, 0, stream>>>(fc2_w, fc2T, 1024, 256);

  ln_f16_kernel<<<Ntok / 8, 256, 0, stream>>>(x, n1g, n1b, h16);
  gemm_wmma_kernel<<<dim3(768 / 128, Ntok / 128), 256, 0, stream>>>(
      h16, qkvT, qkv_b, nullptr, qkvbuf, nullptr, (int)Ntok, 768, 256, 0);
  attn_wmma_kernel<<<2304, 256, 0, stream>>>(qkvbuf, btab, o16);
  gemm_wmma_kernel<<<dim3(256 / 128, Ntok / 128), 256, 0, stream>>>(
      o16, projT, proj_b, x, nullptr, out, (int)Ntok, 256, 256, 1);  // x2 -> d_out
  ln_f16_kernel<<<Ntok / 8, 256, 0, stream>>>(out, n2g, n2b, h16);
  gemm_wmma_kernel<<<dim3(1024 / 128, Ntok / 128), 256, 0, stream>>>(
      h16, fc1T, fc1_b, nullptr, fc1out, nullptr, (int)Ntok, 1024, 256, 2);
  gemm_wmma_kernel<<<dim3(256 / 128, Ntok / 128), 256, 0, stream>>>(
      fc1out, fc2T, fc2_b, out, nullptr, out, (int)Ntok, 256, 1024, 1);
}
